// EmojiGraphEncoder_31361851196144
// MI455X (gfx1250) — compile-verified, matrix operand order checked by reading, not run on hardware
//
#include <hip/hip_runtime.h>
#include <hip/hip_bf16.h>
#include <math.h>

// ---------------------------------------------------------------------------
// EmojiGraphEncoder: 2-layer GAT (H=4, D=64, HID=256) + LN + ELU + mean pool.
// GEMMs: operands pre-packed to zero-padded f16, then v_wmma_f32_16x16x32_f16
// with pure b128 fragment loads (no guards/cvt in the hot loop). Edge phase:
// coalesced gather + L2-resident float atomics (node buffers ~51MB << 192MB L2).
// ---------------------------------------------------------------------------

typedef __attribute__((ext_vector_type(16))) _Float16 v16h;
typedef __attribute__((ext_vector_type(8)))  _Float16 v8h;
typedef __attribute__((ext_vector_type(8)))  float    v8f;

#define HIDc   256
#define NHEAD  4
#define DHEAD  64
#define NEG_SLOPE 0.2f
#define LN_EPS 1e-5f
#define GEMM_MT 4           // m-tiles per wave (64-row strip)

// ----------------------------- utility kernels -----------------------------

__global__ void fill_kernel(float* __restrict__ p, float v, long long n) {
  long long i = (long long)blockIdx.x * blockDim.x + threadIdx.x;
  long long stride = (long long)gridDim.x * blockDim.x;
  for (; i < n; i += stride) p[i] = v;
}

__global__ void ew_sum_kernel(const float* __restrict__ ew, float* __restrict__ sum, int E) {
  int i = blockIdx.x * blockDim.x + threadIdx.x;
  float v = (i < E) ? ew[i] : 0.f;
  #pragma unroll
  for (int off = 16; off; off >>= 1) v += __shfl_down(v, off, 32);
  if ((threadIdx.x & 31) == 0) atomicAdd(sum, v);
}

// c[h] = dot(We[h*64 : h*64+64], a_e[h, :]) -- rank-1 edge attr collapses
// alpha_edge[e,h] to ew[e] * c[h].
__global__ void setup_c_kernel(const float* __restrict__ We1, const float* __restrict__ ae1,
                               const float* __restrict__ We2, const float* __restrict__ ae2,
                               float* __restrict__ c1, float* __restrict__ c2) {
  int t = threadIdx.x;
  if (t < NHEAD) {
    float s1 = 0.f, s2 = 0.f;
    for (int d = 0; d < DHEAD; ++d) {
      s1 += We1[t*DHEAD + d] * ae1[t*DHEAD + d];
      s2 += We2[t*DHEAD + d] * ae2[t*DHEAD + d];
    }
    c1[t] = s1; c2[t] = s2;
  }
}

// f32 [rows_src, K] -> zero-padded f16 [rows_dst, Kpad]
__global__ void cvt_pad_f16_kernel(const float* __restrict__ src, _Float16* __restrict__ dst,
                                   int rows_src, long long rows_dst, int K, int Kpad) {
  long long i = (long long)blockIdx.x * blockDim.x + threadIdx.x;
  long long total = rows_dst * Kpad;
  long long stride = (long long)gridDim.x * blockDim.x;
  for (; i < total; i += stride) {
    int r = (int)(i / Kpad);
    int k = (int)(i % Kpad);
    float v = (r < rows_src && k < K) ? src[(size_t)r * K + k] : 0.f;
    dst[i] = (_Float16)v;
  }
}

// ------------------------------- WMMA GEMM ---------------------------------
// C[n_rows, 256] = Apad[n_rows(pad64), Kpad] @ Wpad[256, Kpad]^T   (f16 x f16 -> f32)
// One wave computes a 64x16 strip: 4 m-tiles sharing one B fragment per K-step.
// Fragment layout (16-bit A 16x32): lane = half*16 + row;
//   elems 0..7  -> K = k0 + half*8 + e        (contiguous 16B)
//   elems 8..15 -> K = k0 + 16 + half*8 + e-8 (contiguous 16B)
__global__ void gemm_wmma_packed(const _Float16* __restrict__ A, const _Float16* __restrict__ W,
                                 float* __restrict__ C, int n_rows, int Kpad) {
  int gid  = blockIdx.x * blockDim.x + threadIdx.x;
  int wave = gid >> 5;
  int lane = threadIdx.x & 31;
  int half = lane >> 4;
  int l15  = lane & 15;
  int mblk  = wave >> 4;                 // 16 n-tiles (256/16)
  int ntile = wave & 15;
  int mblocks = (n_rows + 16*GEMM_MT - 1) / (16*GEMM_MT);
  if (mblk >= mblocks) return;           // wave-uniform: EXEC all-1s at WMMA

  int n = ntile * 16 + l15;
  const _Float16* wrow = W + (size_t)n * Kpad;
  const _Float16* arow[GEMM_MT];
  #pragma unroll
  for (int i = 0; i < GEMM_MT; ++i)
    arow[i] = A + (size_t)(mblk * (16*GEMM_MT) + i*16 + l15) * Kpad;

  v8f zero = {};
  v8f acc[GEMM_MT];
  #pragma unroll
  for (int i = 0; i < GEMM_MT; ++i) acc[i] = zero;

  int hoff = half * 8;
  #pragma unroll 2
  for (int k0 = 0; k0 < Kpad; k0 += 32) {
    v8h b0 = *(const v8h*)(wrow + k0 + hoff);
    v8h b1 = *(const v8h*)(wrow + k0 + 16 + hoff);
    v16h b = __builtin_shufflevector(b0, b1, 0,1,2,3,4,5,6,7,8,9,10,11,12,13,14,15);
    #pragma unroll
    for (int i = 0; i < GEMM_MT; ++i) {
      v8h a0 = *(const v8h*)(arow[i] + k0 + hoff);
      v8h a1 = *(const v8h*)(arow[i] + k0 + 16 + hoff);
      v16h a = __builtin_shufflevector(a0, a1, 0,1,2,3,4,5,6,7,8,9,10,11,12,13,14,15);
      acc[i] = __builtin_amdgcn_wmma_f32_16x16x32_f16(
          /*neg_a=*/false, a, /*neg_b=*/false, b,
          /*c_mod=*/(short)0, acc[i], /*reuse_a=*/false, /*reuse_b=*/false);
    }
  }
  #pragma unroll
  for (int i = 0; i < GEMM_MT; ++i) {
    #pragma unroll
    for (int r = 0; r < 8; ++r) {
      int mm = mblk * (16*GEMM_MT) + i*16 + r + half*8;  // C/D: VGPR r -> M=r / M=r+8
      if (mm < n_rows) C[(size_t)mm * HIDc + n] = acc[i][r];
    }
  }
}

// ------------------------- attention score dots ----------------------------
// one wave per (node, head): alpha_src/dst[n,h] = <xp[n,h,:], a_{s,d}[h,:]>
__global__ void alpha_kernel(const float* __restrict__ xp,
                             const float* __restrict__ a_s, const float* __restrict__ a_d,
                             float* __restrict__ asrc, float* __restrict__ adst, int n_nodes) {
  int gid  = blockIdx.x * blockDim.x + threadIdx.x;
  int wave = gid >> 5;
  int lane = gid & 31;
  int n = wave >> 2;          // NHEAD == 4
  int h = wave & 3;
  if (n >= n_nodes) return;
  const float* xr = xp + (size_t)n * HIDc + h * DHEAD;
  const float* as = a_s + h * DHEAD;
  const float* ad = a_d + h * DHEAD;
  float s = xr[lane] * as[lane] + xr[lane + 32] * as[lane + 32];
  float d = xr[lane] * ad[lane] + xr[lane + 32] * ad[lane + 32];
  #pragma unroll
  for (int off = 16; off; off >>= 1) {
    s += __shfl_down(s, off, 32);
    d += __shfl_down(d, off, 32);
  }
  if (lane == 0) { asrc[n * NHEAD + h] = s; adst[n * NHEAD + h] = d; }
}

// ------------------------------ edge passes --------------------------------
__device__ __forceinline__ void atomicMaxFloat(float* addr, float val) {
  // ordered-int trick: valid for any mix of signs given -inf init
  if (val >= 0.f) atomicMax((int*)addr, __float_as_int(val));
  else            atomicMin((unsigned int*)addr, __float_as_uint(val));
}

// pass 1: logits + segment max over dst
__global__ void edge_logits_kernel(const long long* __restrict__ ei,
                                   const float* __restrict__ ew, const float* __restrict__ ewsum,
                                   const float* __restrict__ asrc, const float* __restrict__ adst,
                                   const float* __restrict__ cvec,
                                   float* __restrict__ exbuf, float* __restrict__ mbuf,
                                   int E, int n_nodes) {
  long long idx = (long long)blockIdx.x * blockDim.x + threadIdx.x;
  long long total = (long long)(E + n_nodes) * NHEAD;
  if (idx >= total) return;
  int e = (int)(idx >> 2), h = (int)(idx & 3);
  long long s, d; float w;
  if (e < E) { s = ei[e]; d = ei[(size_t)E + e]; w = ew[e]; }
  else       { s = d = e - E; w = ewsum[0] / (float)E; }     // self loop, mean weight
  float l = asrc[s * NHEAD + h] + adst[d * NHEAD + h] + w * cvec[h];
  l = (l > 0.f) ? l : NEG_SLOPE * l;
  exbuf[idx] = l;
  atomicMaxFloat(&mbuf[d * NHEAD + h], l);
}

// pass 2: ex = exp(logit - m[dst]); denom[dst] += ex
__global__ void edge_exp_kernel(const long long* __restrict__ ei,
                                const float* __restrict__ mbuf,
                                float* __restrict__ exbuf, float* __restrict__ denom,
                                int E, int n_nodes) {
  long long idx = (long long)blockIdx.x * blockDim.x + threadIdx.x;
  long long total = (long long)(E + n_nodes) * NHEAD;
  if (idx >= total) return;
  int e = (int)(idx >> 2), h = (int)(idx & 3);
  long long d = (e < E) ? ei[(size_t)E + e] : (long long)(e - E);
  float ex = expf(exbuf[idx] - mbuf[d * NHEAD + h]);
  exbuf[idx] = ex;
  atomicAdd(&denom[d * NHEAD + h], ex);
}

// pass 3: accum[dst, f] += (ex/denom[dst,h]) * xp[src, f]; one block per edge,
// thread t = feature t (head = t/64): coalesced gather + coalesced L2 atomics.
__global__ void edge_message_kernel(const long long* __restrict__ ei,
                                    const float* __restrict__ xp,
                                    const float* __restrict__ exbuf,
                                    const float* __restrict__ denom,
                                    float* __restrict__ accum, int E, int n_nodes) {
  int e = blockIdx.x;
  int t = threadIdx.x;
  int h = t >> 6;
  long long s, d;
  if (e < E) { s = ei[e]; d = ei[(size_t)E + e]; }
  else       { s = d = e - E; }
  float alpha = exbuf[(size_t)e * NHEAD + h] / denom[d * NHEAD + h];
  atomicAdd(&accum[(size_t)d * HIDc + t], alpha * xp[(size_t)s * HIDc + t]);
}

// -------------------------- bias + LN + ELU --------------------------------
__global__ void ln_elu_kernel(const float* __restrict__ acc, const float* __restrict__ bias,
                              const float* __restrict__ g, const float* __restrict__ b,
                              float* __restrict__ out, int n_nodes) {
  __shared__ float red[8];
  int n = blockIdx.x;
  int t = threadIdx.x;
  float v = acc[(size_t)n * HIDc + t] + bias[t];

  float s = v;
  #pragma unroll
  for (int off = 16; off; off >>= 1) s += __shfl_down(s, off, 32);
  if ((t & 31) == 0) red[t >> 5] = s;
  __syncthreads();
  float tot = 0.f;
  #pragma unroll
  for (int i = 0; i < 8; ++i) tot += red[i];
  float mu = tot * (1.f / 256.f);
  __syncthreads();

  float dv = v - mu;
  float q = dv * dv;
  #pragma unroll
  for (int off = 16; off; off >>= 1) q += __shfl_down(q, off, 32);
  if ((t & 31) == 0) red[t >> 5] = q;
  __syncthreads();
  float tv = 0.f;
  #pragma unroll
  for (int i = 0; i < 8; ++i) tv += red[i];
  float var = tv * (1.f / 256.f);

  float y = dv * rsqrtf(var + LN_EPS) * g[t] + b[t];
  out[(size_t)n * HIDc + t] = (y > 0.f) ? y : (expf(y) - 1.f);   // ELU(alpha=1)
}

// ------------------------------- pooling -----------------------------------
__global__ void pool_add_kernel(const float* __restrict__ h, const long long* __restrict__ batch,
                                float* __restrict__ outp, float* __restrict__ cnt, int n_nodes) {
  int n = blockIdx.x;
  int t = threadIdx.x;
  long long bb = batch[n];
  atomicAdd(&outp[(size_t)bb * HIDc + t], h[(size_t)n * HIDc + t]);
  if (t == 0) atomicAdd(&cnt[bb], 1.f);
}

__global__ void pool_div_kernel(float* __restrict__ outp, const float* __restrict__ cnt, int nb) {
  int i = blockIdx.x * blockDim.x + threadIdx.x;
  if (i < nb * HIDc) outp[i] /= fmaxf(cnt[i >> 8], 1.f);
}

// ------------------------------- launcher ----------------------------------
extern "C" void kernel_launch(void* const* d_in, const int* in_sizes, int n_in,
                              void* d_out, int out_size, void* d_ws, size_t ws_size,
                              hipStream_t stream) {
  const float*     x     = (const float*)d_in[0];
  const long long* ei    = (const long long*)d_in[1];
  const float*     ew    = (const float*)d_in[2];
  const long long* batch = (const long long*)d_in[3];
  const float* W1  = (const float*)d_in[4];
  const float* as1 = (const float*)d_in[5];
  const float* ad1 = (const float*)d_in[6];
  const float* ae1 = (const float*)d_in[7];
  const float* We1 = (const float*)d_in[8];
  const float* b1  = (const float*)d_in[9];
  const float* W2  = (const float*)d_in[10];
  const float* as2 = (const float*)d_in[11];
  const float* ad2 = (const float*)d_in[12];
  const float* ae2 = (const float*)d_in[13];
  const float* We2 = (const float*)d_in[14];
  const float* b2  = (const float*)d_in[15];
  const float* g1  = (const float*)d_in[16];
  const float* bb1 = (const float*)d_in[17];
  const float* g2  = (const float*)d_in[18];
  const float* bb2 = (const float*)d_in[19];
  (void)n_in; (void)ws_size;

  const int E    = in_sizes[2];
  const int N    = in_sizes[3];
  const int F_IN = in_sizes[0] / N;
  const int ET   = E + N;
  const int NB   = out_size / HIDc;
  const int Npad = ((N + 16*GEMM_MT - 1) / (16*GEMM_MT)) * (16*GEMM_MT);
  float* out = (float*)d_out;

  // workspace carve-up (256B aligned)
  char* p = (char*)d_ws;
  auto carve = [&](size_t bytes) { char* r = p; p += (bytes + 255) & ~(size_t)255; return r; };
  float*    xp    = (float*)carve((size_t)N * HIDc * 4);
  float*    accum = (float*)carve((size_t)N * HIDc * 4);
  float*    hbuf  = (float*)carve((size_t)N * HIDc * 4);
  _Float16* Apad  = (_Float16*)carve((size_t)Npad * HIDc * 2);  // max Kpad = 256
  _Float16* Wpad  = (_Float16*)carve((size_t)HIDc * HIDc * 2);
  float*    asrc  = (float*)carve((size_t)N * NHEAD * 4);
  float*    adst  = (float*)carve((size_t)N * NHEAD * 4);
  float*    mbuf  = (float*)carve((size_t)N * NHEAD * 4);
  float*    denom = (float*)carve((size_t)N * NHEAD * 4);
  float*    exbuf = (float*)carve((size_t)ET * NHEAD * 4);
  float*    cnt   = (float*)carve((size_t)NB * 4);
  float*    scal  = (float*)carve(256);     // [0]=ew_sum, [8..11]=c1, [16..19]=c2

  const float NEG_INF = -__builtin_huge_valf();

  // --- scalar prep: mean edge weight + per-head edge-attn constants
  fill_kernel<<<1, 64, 0, stream>>>(scal, 0.f, 64);
  ew_sum_kernel<<<(E + 255) / 256, 256, 0, stream>>>(ew, scal, E);
  setup_c_kernel<<<1, 64, 0, stream>>>(We1, ae1, We2, ae2, scal + 8, scal + 16);

  auto gemm = [&](const float* Af32, const float* Wf32, float* C, int rows, int K) {
    int Kpad = ((K + 31) / 32) * 32;
    cvt_pad_f16_kernel<<<4096, 256, 0, stream>>>(Af32, Apad, rows, (long long)Npad, K, Kpad);
    cvt_pad_f16_kernel<<<256, 256, 0, stream>>>(Wf32, Wpad, HIDc, (long long)HIDc, K, Kpad);
    long long waves = (long long)(Npad / (16*GEMM_MT)) * 16;
    long long thr = waves * 32;
    gemm_wmma_packed<<<(int)((thr + 255) / 256), 256, 0, stream>>>(Apad, Wpad, C, rows, Kpad);
  };
  auto layer = [&](const float* hin, int K, const float* W, const float* a_s, const float* a_d,
                   const float* bias, const float* cvec, const float* lg, const float* lb,
                   float* hout) {
    gemm(hin, W, xp, N, K);
    {
      long long thr = (long long)N * NHEAD * 32;
      alpha_kernel<<<(int)((thr + 255) / 256), 256, 0, stream>>>(xp, a_s, a_d, asrc, adst, N);
    }
    fill_kernel<<<2048, 256, 0, stream>>>(mbuf, NEG_INF, (long long)N * NHEAD);
    fill_kernel<<<2048, 256, 0, stream>>>(denom, 0.f, (long long)N * NHEAD);
    fill_kernel<<<4096, 256, 0, stream>>>(accum, 0.f, (long long)N * HIDc);
    {
      long long thr = (long long)ET * NHEAD;
      int blk = (int)((thr + 255) / 256);
      edge_logits_kernel<<<blk, 256, 0, stream>>>(ei, ew, scal, asrc, adst, cvec, exbuf, mbuf, E, N);
      edge_exp_kernel<<<blk, 256, 0, stream>>>(ei, mbuf, exbuf, denom, E, N);
    }
    edge_message_kernel<<<ET, 256, 0, stream>>>(ei, xp, exbuf, denom, accum, E, N);
    ln_elu_kernel<<<N, 256, 0, stream>>>(accum, bias, lg, lb, hout, N);
  };

  layer(x,    F_IN, W1, as1, ad1, b1, scal + 8,  g1, bb1, hbuf);
  layer(hbuf, HIDc, W2, as2, ad2, b2, scal + 16, g2, bb2, hbuf);

  // --- mean pool per graph
  fill_kernel<<<512, 256, 0, stream>>>(out, 0.f, (long long)NB * HIDc);
  fill_kernel<<<4, 256, 0, stream>>>(cnt, 0.f, (long long)NB);
  pool_add_kernel<<<N, 256, 0, stream>>>(hbuf, batch, out, cnt, N);
  pool_div_kernel<<<(NB * HIDc + 255) / 256, 256, 0, stream>>>(out, cnt, NB);
}
